// RelationshipFusionWithAttention_90537910600027
// MI455X (gfx1250) — matrix operand order repaired and checked by reading, not compiled, not run on hardware
//
#include <hip/hip_runtime.h>
#include <hip/hip_bf16.h>
#include <math.h>

typedef __attribute__((ext_vector_type(16))) _Float16 v16h;
typedef __attribute__((ext_vector_type(8)))  _Float16 v8h;
typedef __attribute__((ext_vector_type(8)))  float    v8f;

#define B_ 8
#define N_ 128
#define D_ 256
#define R_ 64

__device__ __forceinline__ float gelu_exact(float v) {
  return 0.5f * v * (1.0f + erff(v * 0.70710678118654752440f));
}

// A fragment (16x32 f16, row-major source): lane l: M = l&15, g = l>>4.
// halves 0..7 -> K = kb + g*8 + h ; halves 8..15 -> K = kb + 16 + g*8 + (h-8)
__device__ __forceinline__ v16h frag_a(const _Float16* rowptr, int kb, int g) {
  v8h lo = *(const v8h*)(rowptr + kb + g * 8);
  v8h hi = *(const v8h*)(rowptr + kb + 16 + g * 8);
  v16h a;
#pragma unroll
  for (int t = 0; t < 8; ++t) { a[t] = lo[t]; a[t + 8] = hi[t]; }
  return a;
}

// B fragment (32x16 f16) from transposed (NxK row-major) weights:
// lane l: N = l&15, g = l>>4 ; halves 0..7 -> K = kb + g*8 + h ; halves 8..15 -> K = kb + 16 + g*8 + (h-8)
__device__ __forceinline__ v16h frag_b(const _Float16* colptr, int kb, int g) {
  v8h lo = *(const v8h*)(colptr + kb + g * 8);
  v8h hi = *(const v8h*)(colptr + kb + 16 + g * 8);
  v16h b;
#pragma unroll
  for (int t = 0; t < 8; ++t) { b[t] = lo[t]; b[t + 8] = hi[t]; }
  return b;
}

__device__ __forceinline__ v8f wmma16(v16h a, v16h b, v8f c) {
  return __builtin_amdgcn_wmma_f32_16x16x32_f16(false, a, false, b, (short)0, c,
                                                false, false);
}

// ---------------- converters ----------------

__global__ __launch_bounds__(256) void cvt_f16_kernel(const float* __restrict__ s,
                                                      _Float16* __restrict__ d, int n) {
  int t = blockIdx.x * 256 + threadIdx.x;
  if (t < n) d[t] = (_Float16)s[t];
}

// dst (N x K, f16) = transpose of src rows [rowoff, rowoff+K) x N (f32)
__global__ __launch_bounds__(256) void transpose_f16_kernel(const float* __restrict__ s,
                                                            _Float16* __restrict__ d,
                                                            int rowoff, int K, int N) {
  int t = blockIdx.x * 256 + threadIdx.x;
  if (t < K * N) {
    int n = t / K, k = t % K;
    d[t] = (_Float16)s[(size_t)(rowoff + k) * N + n];
  }
}

// ---------------- generic WMMA GEMM: C = A(MxK) * Bt^T + bias (+resid) ----------------
// one wave per 16x16 output tile
__global__ __launch_bounds__(32) void wmma_gemm16(
    const _Float16* __restrict__ A,   // M x K row-major (f16)
    const _Float16* __restrict__ Bt,  // N x K row-major (B transposed, f16)
    const float* __restrict__ bias,   // N or nullptr
    const float* __restrict__ resid,  // M x N or nullptr
    float* __restrict__ Cf,           // M x N or nullptr
    _Float16* __restrict__ Ch,        // M x N or nullptr
    int M, int N, int K) {
  const int mt = blockIdx.x;
  const int nt = blockIdx.y;
  const int lane = threadIdx.x;
  const int g = lane >> 4;
  const int lr = lane & 15;
  const _Float16* arow = A + (size_t)(mt * 16 + lr) * K;
  const _Float16* bcol = Bt + (size_t)(nt * 16 + lr) * K;
  v8f acc = {};
  for (int kb = 0; kb < K; kb += 32)
    acc = wmma16(frag_a(arow, kb, g), frag_b(bcol, kb, g), acc);
  const int col = nt * 16 + lr;
  const float bv = bias ? bias[col] : 0.0f;
#pragma unroll
  for (int p = 0; p < 8; ++p) {
    const int row = mt * 16 + p + g * 8;  // C layout: vgpr p, lane half g
    float v = acc[p] + bv;
    if (resid) v += resid[(size_t)row * N + col];
    if (Cf) Cf[(size_t)row * N + col] = v;
    if (Ch) Ch[(size_t)row * N + col] = (_Float16)v;
  }
}

// ---------------- fused per-(b,i) edge-row kernel ----------------
// grid = B*N blocks, 256 threads (8 waves). Wave w owns j-block [16w,16w+16).
__global__ __launch_bounds__(256) void fuse_attn_kernel(
    const float* __restrict__ Rel,        // B,N,N,R f32
    const float* __restrict__ xjm,        // (B*N) x D : x@W1m[:D]+b1m
    const float* __restrict__ xia,        // (B*N) x D : x@W1a[:D]+b1a
    const float* __restrict__ xja,        // (B*N) x D : x@W1a[D:2D]
    const _Float16* __restrict__ WmRelT,  // D x R (transposed f16)
    const _Float16* __restrict__ WaRelT,  // D x R (transposed f16)
    const float* __restrict__ W2a,        // D
    _Float16* __restrict__ hbar) {        // (B*N) x D output (f16)
  __shared__ __align__(16) _Float16 rel_h[N_ * R_];  // 16 KB
  __shared__ __align__(16) _Float16 h_lds[N_ * D_];  // 64 KB
  __shared__ float e_lds[N_];
  __shared__ float alpha[N_];

  const int bi = blockIdx.x;
  const int b = bi >> 7;
  const int i = bi & (N_ - 1);
  const int tid = threadIdx.x;

  // stage Rel[b,i,:,:] -> LDS f16
  const float* relp = Rel + (size_t)bi * (N_ * R_);
  for (int t = tid * 4; t < N_ * R_; t += 256 * 4) {
    float4 v = *(const float4*)(relp + t);
    rel_h[t + 0] = (_Float16)v.x;
    rel_h[t + 1] = (_Float16)v.y;
    rel_h[t + 2] = (_Float16)v.z;
    rel_h[t + 3] = (_Float16)v.w;
  }
  __syncthreads();

  const int wave = tid >> 5;
  const int lane = tid & 31;
  const int g = lane >> 4;
  const int lr = lane & 15;
  const int mbase = wave * 16;
  const size_t xb = (size_t)b * N_ * D_;
  const _Float16* arow = &rel_h[(mbase + lr) * R_];

  v8f eacc = {};
  for (int dt = 0; dt < D_ / 16; ++dt) {
    const int col = dt * 16 + lr;
    const _Float16* bm = WmRelT + (size_t)col * R_;
    const _Float16* ba = WaRelT + (size_t)col * R_;
    v8f cm = {}, ca = {};
#pragma unroll
    for (int ks = 0; ks < 2; ++ks) {
      v16h a = frag_a(arow, ks * 32, g);      // shared A for both paths
      cm = wmma16(a, frag_b(bm, ks * 32, g), cm);
      ca = wmma16(a, frag_b(ba, ks * 32, g), ca);
    }
    const float xiav = xia[xb + (size_t)i * D_ + col];
    const float w2av = W2a[col];
#pragma unroll
    for (int p = 0; p < 8; ++p) {
      const int j = mbase + p + g * 8;
      const float hm = gelu_exact(cm[p] + xjm[xb + (size_t)j * D_ + col]);
      h_lds[j * D_ + col] = (_Float16)hm;
      const float ha = gelu_exact(ca[p] + xiav + xja[xb + (size_t)j * D_ + col]);
      eacc[p] += ha * w2av;
    }
  }
  // reduce logits across the 16 columns held by each half-wave
#pragma unroll
  for (int m = 8; m >= 1; m >>= 1) {
#pragma unroll
    for (int p = 0; p < 8; ++p) eacc[p] += __shfl_xor(eacc[p], m, 32);
  }
  if (lr == 0) {
#pragma unroll
    for (int p = 0; p < 8; ++p) e_lds[mbase + p + g * 8] = eacc[p];
  }
  __syncthreads();

  // softmax over j (diagonal masked), single wave
  if (tid < 32) {
    float ev[4];
    float mx = -3.0e38f;
#pragma unroll
    for (int q = 0; q < 4; ++q) {
      const int j = tid + q * 32;
      const float e = (j == i) ? -3.0e38f : e_lds[j];
      ev[q] = e;
      mx = fmaxf(mx, e);
    }
#pragma unroll
    for (int m = 16; m >= 1; m >>= 1) mx = fmaxf(mx, __shfl_xor(mx, m, 32));
    float s = 0.0f;
    float pv[4];
#pragma unroll
    for (int q = 0; q < 4; ++q) {
      const int j = tid + q * 32;
      pv[q] = (j == i) ? 0.0f : __expf(ev[q] - mx);
      s += pv[q];
    }
#pragma unroll
    for (int m = 16; m >= 1; m >>= 1) s += __shfl_xor(s, m, 32);
    const float inv = 1.0f / s;
#pragma unroll
    for (int q = 0; q < 4; ++q) alpha[tid + q * 32] = pv[q] * inv;
  }
  __syncthreads();

  // hbar[d] = sum_j alpha[j] * h[j,d]  (softmax sums to 1 => W2m/b2m folded later)
  {
    const int d = tid;
    float acc = 0.0f;
    for (int j = 0; j < N_; ++j) acc += alpha[j] * (float)h_lds[j * D_ + d];
    hbar[(size_t)bi * D_ + d] = (_Float16)acc;
  }
}

// ---------------- launch ----------------

extern "C" void kernel_launch(void* const* d_in, const int* in_sizes, int n_in,
                              void* d_out, int out_size, void* d_ws, size_t ws_size,
                              hipStream_t stream) {
  (void)in_sizes; (void)n_in; (void)out_size; (void)ws_size;
  const float* x   = (const float*)d_in[0];  // (B,N,D)
  const float* Rel = (const float*)d_in[1];  // (B,N,N,R)
  const float* W1m = (const float*)d_in[2];  // (D+R, D)
  const float* b1m = (const float*)d_in[3];
  const float* W2m = (const float*)d_in[4];  // (D, D)
  const float* b2m = (const float*)d_in[5];
  const float* W1a = (const float*)d_in[6];  // (2D+R, D)
  const float* b1a = (const float*)d_in[7];
  const float* W2a = (const float*)d_in[8];  // (D, 1)
  // d_in[9] = b2a: constant over j, cancels in softmax.

  const int MN = B_ * N_;  // 1024 rows

  char* w = (char*)d_ws;
  auto alloc = [&](size_t bytes) { void* p = (void*)w; w += (bytes + 255) & ~(size_t)255; return p; };
  _Float16* x_h    = (_Float16*)alloc((size_t)MN * D_ * 2);
  _Float16* W1mxT  = (_Float16*)alloc((size_t)D_ * D_ * 2);
  _Float16* W1axiT = (_Float16*)alloc((size_t)D_ * D_ * 2);
  _Float16* W1axjT = (_Float16*)alloc((size_t)D_ * D_ * 2);
  _Float16* WmRelT = (_Float16*)alloc((size_t)D_ * R_ * 2);
  _Float16* WaRelT = (_Float16*)alloc((size_t)D_ * R_ * 2);
  _Float16* W2mT   = (_Float16*)alloc((size_t)D_ * D_ * 2);
  float*    xjm    = (float*)alloc((size_t)MN * D_ * 4);
  float*    xia    = (float*)alloc((size_t)MN * D_ * 4);
  float*    xja    = (float*)alloc((size_t)MN * D_ * 4);
  _Float16* hbar   = (_Float16*)alloc((size_t)MN * D_ * 2);

  // 1) precision conversion / weight transposes
  cvt_f16_kernel<<<(MN * D_ + 255) / 256, 256, 0, stream>>>(x, x_h, MN * D_);
  transpose_f16_kernel<<<(D_ * D_ + 255) / 256, 256, 0, stream>>>(W1m, W1mxT, 0, D_, D_);
  transpose_f16_kernel<<<(R_ * D_ + 255) / 256, 256, 0, stream>>>(W1m, WmRelT, D_, R_, D_);
  transpose_f16_kernel<<<(D_ * D_ + 255) / 256, 256, 0, stream>>>(W1a, W1axiT, 0, D_, D_);
  transpose_f16_kernel<<<(D_ * D_ + 255) / 256, 256, 0, stream>>>(W1a, W1axjT, D_, D_, D_);
  transpose_f16_kernel<<<(R_ * D_ + 255) / 256, 256, 0, stream>>>(W1a, WaRelT, 2 * D_, R_, D_);
  transpose_f16_kernel<<<(D_ * D_ + 255) / 256, 256, 0, stream>>>(W2m, W2mT, 0, D_, D_);

  // 2) node projections (WMMA GEMMs, 1024x256x256)
  dim3 ggrid(MN / 16, D_ / 16);
  wmma_gemm16<<<ggrid, 32, 0, stream>>>(x_h, W1mxT,  b1m, nullptr, xjm, nullptr, MN, D_, D_);
  wmma_gemm16<<<ggrid, 32, 0, stream>>>(x_h, W1axiT, b1a, nullptr, xia, nullptr, MN, D_, D_);
  wmma_gemm16<<<ggrid, 32, 0, stream>>>(x_h, W1axjT, nullptr, nullptr, xja, nullptr, MN, D_, D_);

  // 3) fused edge GEMMs + GELU + attention logits + softmax + alpha-weighted h-sum
  fuse_attn_kernel<<<MN, 256, 0, stream>>>(Rel, xjm, xia, xja, WmRelT, WaRelT, W2a, hbar);

  // 4) out = x + hbar @ W2m + b2m  (WMMA GEMM with bias + residual)
  wmma_gemm16<<<ggrid, 32, 0, stream>>>(hbar, W2mT, b2m, x, (float*)d_out, nullptr, MN, D_, D_);
}
